// LSTM_32169305047239
// MI455X (gfx1250) — matrix operand order repaired and checked by reading, not compiled, not run on hardware
//
#include <hip/hip_runtime.h>
#include <math.h>

typedef __attribute__((ext_vector_type(16))) _Float16 v16h;
typedef __attribute__((ext_vector_type(8)))  _Float16 v8h;
typedef __attribute__((ext_vector_type(8)))  float    v8f;

#define BATCH  2048
#define TSTEPS 2048
#define NIN    5
#define NH     64
#define NOUT   10
#define BB     16       // batch rows per block
#define NBLK   (BATCH / BB)
#define NTHR   128      // 4 waves
#define XK     32       // x^T staging tile K extent: cols 0..4 = x_t, col 5 = 1.0 (bias), rest 0

static __device__ __forceinline__ float fast_tanh(float x) {
#if __has_builtin(__builtin_amdgcn_tanhf)
  return __builtin_amdgcn_tanhf(x);            // v_tanh_f32 (gfx1250 TRANS)
#else
  const float e = __builtin_amdgcn_exp2f(2.8853900817779268f * x); // e^(2x)
  return 1.0f - 2.0f * __builtin_amdgcn_rcpf(e + 1.0f);
#endif
}
static __device__ __forceinline__ float fast_sigmoid(float x) {
#if __has_builtin(__builtin_amdgcn_tanhf)
  return __builtin_fmaf(0.5f, __builtin_amdgcn_tanhf(0.5f * x), 0.5f); // 1 TRANS
#else
  return __builtin_amdgcn_rcpf(1.0f + __builtin_amdgcn_exp2f(-1.4426950408889634f * x));
#endif
}

// GEMM is computed transposed: G^T[4H x 16] = [W_hh | W_ih | b] * [h | x | 1]^T
//  - A operand (16x32 f16, contiguous k = 16*hi + s): weight tiles, CONSTANT -> registers
//  - B operand (32x16 f16, same contiguous k = 16*hi + s): h^T / x^T chunk = 32 consecutive bytes
//    of the row-major [16][64] h tile per lane -> ds_load_b128
//  - C (16x16 f32): lane lm owns batch row m=lm, VGPR v owns gate-row v+8*hi
//    -> each lane gets 8 consecutive h columns -> single ds_store_b128 write-back
__global__ __launch_bounds__(NTHR)
void lstm_fused_kernel(const float* __restrict__ x,
                       const int*   __restrict__ ytgt,
                       const float* __restrict__ W_ih,
                       const float* __restrict__ W_hh,
                       const float* __restrict__ b_ih,
                       const float* __restrict__ b_hh,
                       const float* __restrict__ W1,
                       const float* __restrict__ b1,
                       const float* __restrict__ W2,
                       const float* __restrict__ b2,
                       float* __restrict__ out,          // logits [BATCH][NOUT]
                       float* __restrict__ loss_partial) // [NBLK]
{
  __shared__ _Float16 h_lds[2][BB * NH];   // double-buffered h^T source (row-major [m][k])
  __shared__ _Float16 x_a[2][BB * XK];     // double-buffered [x_t | 1 | 0...] tile, f16
  __shared__ float    h32[BB * NH];        // final h, f32
  __shared__ float    z_lds[BB * NH];
  __shared__ float    logit_lds[BB * NOUT];
  __shared__ float    part[BB];

  const int tid  = threadIdx.x;
  const int j    = tid >> 5;        // wave 0..3 -> M-tile set {j, j+4, j+8, j+12} (gate g tile = g*4+j)
  const int lane = tid & 31;
  const int lm   = lane & 15;       // A row within tile / B,C column (batch row m)
  const int hi   = lane >> 4;
  const int r0   = blockIdx.x * BB;
  const int hb0  = j * 16 + 8 * hi; // first of the 8 h-columns this lane owns

  // zero h buffer 0; init x_a: zeros with col 5 = 1.0 (bias multiplier), both buffers
  for (int idx = tid; idx < BB * NH; idx += NTHR) h_lds[0][idx] = (_Float16)0.0f;
  for (int idx = tid; idx < 2 * BB * XK; idx += NTHR)
    (&x_a[0][0])[idx] = (_Float16)(((idx & (XK - 1)) == NIN) ? 1.0f : 0.0f);

  // ---- constant A-fragments (16x32, contiguous k = 16*hi + s), whole loop ----
  v16h af[4][2];   // W_hh tiles, K chunks 0..63
  v16h ax[4];      // [W_ih | bias | 0] tile, K chunk for x
#pragma unroll
  for (int g = 0; g < 4; ++g) {
    const int n = g * NH + j * 16 + lm;      // gate row
    const float* wrow = W_hh + n * NH;
    const float bias = b_ih[n] + b_hh[n];
#pragma unroll
    for (int kc = 0; kc < 2; ++kc)
#pragma unroll
      for (int q = 0; q < 8; ++q) {
        const int k = 16 * hi + 2 * q;
        af[g][kc][2 * q]     = (_Float16)wrow[kc * 32 + k];
        af[g][kc][2 * q + 1] = (_Float16)wrow[kc * 32 + k + 1];
      }
#pragma unroll
    for (int q = 0; q < 8; ++q) {
      const int k = 16 * hi + 2 * q;
#pragma unroll
      for (int s = 0; s < 2; ++s) {
        const int kk = k + s;
        const float v = (kk < NIN) ? W_ih[n * NIN + kk] : ((kk == NIN) ? bias : 0.0f);
        ax[g][2 * q + s] = (_Float16)v;
      }
    }
  }

  float c8[8];
#pragma unroll
  for (int v = 0; v < 8; ++v) c8[v] = 0.0f;

  __syncthreads();   // x_a zero/bias-col complete before staging cols 0..4

  // ---- x staging: stage t=0, prefetch t=1 into a register -------------
  const float* xptr = nullptr;
  float xreg = 0.0f;
  const int xm = tid / NIN, xi = tid % NIN;
  if (tid < BB * NIN) {
    xptr = x + ((size_t)(r0 + xm) * TSTEPS) * NIN + xi;
    x_a[0][xm * XK + xi] = (_Float16)xptr[0];
    xreg = xptr[NIN];                 // t = 1
  }
  __syncthreads();

  // ---- time loop: ONE barrier per step --------------------------------
  for (int t = 0; t < TSTEPS; ++t) {
    const int cur = t & 1, nxt = cur ^ 1;
    const _Float16* __restrict__ hsrc = h_lds[cur];
    const _Float16* __restrict__ xsrc = x_a[cur];

    // B fragments: 16 consecutive f16 (32 bytes) per chunk per lane -> ds_load_b128 pairs
    v16h bh0, bh1, bx;
#pragma unroll
    for (int e = 0; e < 16; ++e) {
      bh0[e] = hsrc[lm * NH + 16 * hi + e];
      bh1[e] = hsrc[lm * NH + 32 + 16 * hi + e];
      bx[e]  = xsrc[lm * XK + 16 * hi + e];
    }

    // 4 gate tiles: K = 32(x,1,bias) + 64(h) as 3 chained WMMAs each
    v8f acc[4];
#pragma unroll
    for (int g = 0; g < 4; ++g) {
      v8f a = {};
      a = __builtin_amdgcn_wmma_f32_16x16x32_f16(false, ax[g],    false, bx,
                                                 (short)0, a, false, false);
      a = __builtin_amdgcn_wmma_f32_16x16x32_f16(false, af[g][0], false, bh0,
                                                 (short)0, a, false, false);
      a = __builtin_amdgcn_wmma_f32_16x16x32_f16(false, af[g][1], false, bh1,
                                                 (short)0, a, false, false);
      acc[g] = a;
    }

    // stage x(t+1) into other buffer; prefetch x(t+2) (latency hidden 1 iter)
    if (tid < BB * NIN) {
      x_a[nxt][xm * XK + xi] = (_Float16)xreg;
      const int tp = (t + 2 < TSTEPS) ? (t + 2) : (TSTEPS - 1);
      xreg = xptr[(size_t)tp * NIN];
    }

    // LSTM cell: lane owns (m=lm, h = hb0+v); gates fully in registers
    v8h hp;
#pragma unroll
    for (int v = 0; v < 8; ++v) {
      const float ig = fast_sigmoid(acc[0][v]);
      const float fg = fast_sigmoid(acc[1][v]);
      const float gg = fast_tanh(acc[2][v]);
      const float og = fast_sigmoid(acc[3][v]);
      c8[v] = fg * c8[v] + ig * gg;
      const float hval = og * fast_tanh(c8[v]);
      hp[v] = (_Float16)hval;
      if (t == TSTEPS - 1) h32[lm * NH + hb0 + v] = hval;  // f32 for the head
    }
    *(v8h*)&h_lds[nxt][lm * NH + hb0] = hp;   // single ds_store_b128
    __syncthreads();
  }

  // ---- MLP head: z = tanh(h @ W1^T + b1) ------------------------------
  const int zm  = tid >> 3;
  const int zj0 = (tid & 7) * 8;
#pragma unroll
  for (int q = 0; q < 8; ++q) {
    const int jj = zj0 + q;
    float a = b1[jj];
    const float* w1r = W1 + jj * NH;
    for (int k = 0; k < NH; ++k) a += h32[zm * NH + k] * w1r[k];
    z_lds[zm * NH + jj] = fast_tanh(a);
  }
  __syncthreads();

  // logits = z @ W2^T + b2
  for (int idx = tid; idx < BB * NOUT; idx += NTHR) {
    const int m = idx / NOUT, o = idx % NOUT;
    float a = b2[o];
    const float* w2r = W2 + o * NH;
    for (int jj = 0; jj < NH; ++jj) a += z_lds[m * NH + jj] * w2r[jj];
    logit_lds[idx] = a;
    out[(size_t)(r0 + m) * NOUT + o] = a;
  }
  __syncthreads();

  // per-row NLL, per-block partial (deterministic two-pass reduction)
  if (tid < BB) {
    float mx = logit_lds[tid * NOUT];
    for (int o = 1; o < NOUT; ++o) mx = fmaxf(mx, logit_lds[tid * NOUT + o]);
    float s = 0.0f;
    for (int o = 0; o < NOUT; ++o) s += __expf(logit_lds[tid * NOUT + o] - mx);
    const float lse = __logf(s) + mx;
    const int tgt = ytgt[r0 + tid];
    part[tid] = lse - logit_lds[tid * NOUT + tgt];
  }
  __syncthreads();
  if (tid == 0) {
    float s = 0.0f;
    for (int m = 0; m < BB; ++m) s += part[m];
    loss_partial[blockIdx.x] = s;
  }
}

__global__ void loss_reduce_kernel(const float* __restrict__ partial,
                                   float* __restrict__ out)
{
  float s = 0.0f;
  for (int i = 0; i < NBLK; ++i) s += partial[i];
  out[(size_t)BATCH * NOUT] = s / (float)BATCH;
}

extern "C" void kernel_launch(void* const* d_in, const int* in_sizes, int n_in,
                              void* d_out, int out_size, void* d_ws, size_t ws_size,
                              hipStream_t stream) {
  const float* x     = (const float*)d_in[0];
  const int*   ytgt  = (const int*)  d_in[1];
  const float* W_ih  = (const float*)d_in[2];
  const float* W_hh  = (const float*)d_in[3];
  const float* b_ih  = (const float*)d_in[4];
  const float* b_hh  = (const float*)d_in[5];
  const float* W1    = (const float*)d_in[6];
  const float* b1    = (const float*)d_in[7];
  const float* W2    = (const float*)d_in[8];
  const float* b2    = (const float*)d_in[9];
  float* out = (float*)d_out;
  float* loss_partial = (float*)d_ws;   // NBLK floats of scratch

  lstm_fused_kernel<<<NBLK, NTHR, 0, stream>>>(x, ytgt, W_ih, W_hh, b_ih, b_hh,
                                               W1, b1, W2, b2, out, loss_partial);
  loss_reduce_kernel<<<1, 1, 0, stream>>>(loss_partial, out);
}
